// SupportModulatedAttention_39694087750158
// MI455X (gfx1250) — matrix operand order repaired and checked by reading, not compiled, not run on hardware
//
#include <hip/hip_runtime.h>
#include <hip/hip_bf16.h>

#define BSZ   8
#define CDIM  256
#define NDIM  4096
#define HEADS 8

typedef __bf16 bf16;
typedef bf16  v16bf __attribute__((ext_vector_type(16)));
typedef bf16  v8bf  __attribute__((ext_vector_type(8)));
typedef bf16  v4bf  __attribute__((ext_vector_type(4)));
typedef float v8f   __attribute__((ext_vector_type(8)));

union V16 { v16bf v; v8bf h[2]; };

// ---------------------------------------------------------------------------
// Gate: s = proto @ mod_w^T + mod_b ; LayerNorm ; sigmoid    -> gate[B][256]
// ---------------------------------------------------------------------------
__global__ __launch_bounds__(256) void gate_kernel(
    const float* __restrict__ proto, const float* __restrict__ mod_w,
    const float* __restrict__ mod_b, const float* __restrict__ ln_g,
    const float* __restrict__ ln_b, float* __restrict__ gate)
{
  const int b = blockIdx.x, t = threadIdx.x;
  const float* p = proto + b * CDIM;
  float s = mod_b[t];
  const float* wrow = mod_w + (size_t)t * CDIM;
  for (int c = 0; c < CDIM; ++c) s += p[c] * wrow[c];
  __shared__ float red[256];
  red[t] = s; __syncthreads();
  for (int off = 128; off > 0; off >>= 1) { if (t < off) red[t] += red[t + off]; __syncthreads(); }
  const float mu = red[0] * (1.0f / 256.0f);
  __syncthreads();
  const float d = s - mu;
  red[t] = d * d; __syncthreads();
  for (int off = 128; off > 0; off >>= 1) { if (t < off) red[t] += red[t + off]; __syncthreads(); }
  const float var = red[0] * (1.0f / 256.0f);
  const float ln = d * rsqrtf(var + 1e-5f) * ln_g[t] + ln_b[t];
  gate[b * CDIM + t] = 1.0f / (1.0f + __expf(-ln));
}

// ---------------------------------------------------------------------------
// Fused depthwise: out = dw7x7( SiLU( dw3x3(x) + b3 ) ) + b7, per (b,c) plane
// ---------------------------------------------------------------------------
__global__ __launch_bounds__(256) void dw_fused_kernel(
    const float* __restrict__ x,
    const float* __restrict__ w3, const float* __restrict__ b3,
    const float* __restrict__ w7, const float* __restrict__ b7,
    float* __restrict__ out)
{
  __shared__ float xs[66][66];
  __shared__ float t3[64][64];
  __shared__ float w3s[9], w7s[49];
  const int plane = blockIdx.x;           // b*C + c
  const int c = plane & (CDIM - 1);
  const float* g = x + (size_t)plane * NDIM;
  const int tid = threadIdx.x;
  if (tid < 9)  w3s[tid] = w3[c * 9 + tid];
  if (tid < 49) w7s[tid] = w7[c * 49 + tid];
  const float bias3 = b3[c], bias7 = b7[c];
  for (int i = tid; i < 66 * 66; i += 256) {
    const int yy = i / 66 - 1, xx = i % 66 - 1;
    float v = 0.0f;
    if (yy >= 0 && yy < 64 && xx >= 0 && xx < 64) v = g[yy * 64 + xx];
    xs[i / 66][i % 66] = v;
  }
  __syncthreads();
  for (int i = tid; i < NDIM; i += 256) {
    const int y = i >> 6, xx = i & 63;
    float s = bias3;
    #pragma unroll
    for (int dy = 0; dy < 3; ++dy)
      #pragma unroll
      for (int dx = 0; dx < 3; ++dx)
        s += w3s[dy * 3 + dx] * xs[y + dy][xx + dx];
    t3[y][xx] = s / (1.0f + __expf(-s));     // SiLU
  }
  __syncthreads();
  for (int i = tid; i < NDIM; i += 256) {
    const int y = i >> 6, xx = i & 63;
    float s = bias7;
    for (int dy = -3; dy <= 3; ++dy) {
      const int yy = y + dy;
      if (yy < 0 || yy >= 64) continue;
      for (int dx = -3; dx <= 3; ++dx) {
        const int x2 = xx + dx;
        if (x2 < 0 || x2 >= 64) continue;
        s += w7s[(dy + 3) * 7 + (dx + 3)] * t3[yy][x2];
      }
    }
    out[(size_t)plane * NDIM + i] = s;
  }
}

// ---------------------------------------------------------------------------
// Generic GEMM (1x1 conv): Y[b,o,n] = sum_c W[o,c]*X[b,c,n] + bias[o]
// bf16 WMMA, double-buffered LDS, register-staged vector loads.
// Block tile 128(o) x 128(n); 8 waves, each owning 32o x 64n (8 WMMA tiles).
// ---------------------------------------------------------------------------
__device__ __forceinline__ void gemm_load_regs(
    const float* __restrict__ W, const float* __restrict__ Xb,
    int o0, int n0b, int kc, int tid, float4 aA[4], float bB[16])
{
  #pragma unroll
  for (int r = 0; r < 4; ++r) {                       // A: 128x32 f32, b128 loads
    const int i = tid + 256 * r;
    const int o = i >> 3, c4 = i & 7;
    aA[r] = *(const float4*)&W[(size_t)(o0 + o) * CDIM + kc + c4 * 4];
  }
  const int n = tid & 127, cb = (tid >> 7) * 16;       // B: 16 coalesced b32 loads
  #pragma unroll
  for (int j = 0; j < 16; ++j)
    bB[j] = Xb[(size_t)(kc + cb + j) * NDIM + n0b + n];
}

__device__ __forceinline__ void gemm_store_lds(
    bf16 (*__restrict__ As)[40], bf16 (*__restrict__ Bs)[40],
    int tid, const float4 aA[4], const float bB[16])
{
  #pragma unroll
  for (int r = 0; r < 4; ++r) {
    const int i = tid + 256 * r;
    const int o = i >> 3, c4 = i & 7;
    v4bf t;
    t[0] = (bf16)aA[r].x; t[1] = (bf16)aA[r].y;
    t[2] = (bf16)aA[r].z; t[3] = (bf16)aA[r].w;
    *(v4bf*)&As[o][c4 * 4] = t;                        // ds_store_b64
  }
  const int n = tid & 127, cb = (tid >> 7) * 16;
  V16 t16;
  #pragma unroll
  for (int j = 0; j < 16; ++j) t16.v[j] = (bf16)bB[j];
  *(v8bf*)&Bs[n][cb]     = t16.h[0];                   // ds_store_b128 x2
  *(v8bf*)&Bs[n][cb + 8] = t16.h[1];
}

__global__ __launch_bounds__(256) void gemm_wmma_kernel(
    const float* __restrict__ W, const float* __restrict__ bias,
    const float* __restrict__ X, float* __restrict__ Y)
{
  __shared__ bf16 As[2][128][40];    // [o][c], rows padded to 80B (16B aligned)
  __shared__ bf16 Bs[2][128][40];    // [n][c] transposed
  const int b   = blockIdx.z;
  const int o0  = blockIdx.y * 128;
  const int n0b = blockIdx.x * 128;
  const int tid = threadIdx.x;
  const int wid = tid >> 5, lane = tid & 31;
  const int half = lane >> 4, lr = lane & 15;
  const int wo = wid & 3;            // 32-row strip of the 128 o rows
  const int wn = wid >> 2;           // 64-col half of the 128 n cols
  const float* Xb = X + (size_t)b * CDIM * NDIM;

  v8f acc[2][4] = {};
  float4 aA[4];
  float  bB[16];

  gemm_load_regs(W, Xb, o0, n0b, 0, tid, aA, bB);
  gemm_store_lds(As[0], Bs[0], tid, aA, bB);
  int cur = 0;
  for (int kc = 0; kc < CDIM; kc += 32) {
    __syncthreads();
    const bool has_next = (kc + 32 < CDIM);            // block-uniform branch
    if (has_next) gemm_load_regs(W, Xb, o0, n0b, kc + 32, tid, aA, bB);
    // --- compute from LDS[cur] ---
    V16 a[2];
    #pragma unroll
    for (int mi = 0; mi < 2; ++mi) {
      const bf16* ap = &As[cur][wo * 32 + mi * 16 + lr][half * 8];
      a[mi].h[0] = *(const v8bf*)ap;                   // K 0-7  (or 8-15)
      a[mi].h[1] = *(const v8bf*)(ap + 16);            // K 16-23 (or 24-31)
    }
    #pragma unroll
    for (int t = 0; t < 4; ++t) {
      V16 bf_;
      const bf16* bp = &Bs[cur][wn * 64 + t * 16 + lr][half * 8];
      bf_.h[0] = *(const v8bf*)bp;
      bf_.h[1] = *(const v8bf*)(bp + 16);
      #pragma unroll
      for (int mi = 0; mi < 2; ++mi)
        acc[mi][t] = __builtin_amdgcn_wmma_f32_16x16x32_bf16(
            false, a[mi].v, false, bf_.v, (short)0, acc[mi][t], false, false);
    }
    if (has_next) gemm_store_lds(As[cur ^ 1], Bs[cur ^ 1], tid, aA, bB);
    cur ^= 1;
  }
  #pragma unroll
  for (int mi = 0; mi < 2; ++mi) {
    #pragma unroll
    for (int t = 0; t < 4; ++t) {
      const int n = n0b + wn * 64 + t * 16 + lr;
      #pragma unroll
      for (int r = 0; r < 8; ++r) {
        const int o = o0 + wo * 32 + mi * 16 + half * 8 + r;
        Y[((size_t)b * CDIM + o) * NDIM + n] = acc[mi][t][r] + bias[o];
      }
    }
  }
}

// ---------------------------------------------------------------------------
// Softmax over spatial axis (rows of length 4096), in place
// ---------------------------------------------------------------------------
__global__ __launch_bounds__(256) void softmax_rows_kernel(float* __restrict__ k)
{
  float* row = k + (size_t)blockIdx.x * NDIM;
  const int t = threadIdx.x;
  __shared__ float red[256];
  float m = -1e30f;
  for (int i = t; i < NDIM; i += 256) m = fmaxf(m, row[i]);
  red[t] = m; __syncthreads();
  for (int off = 128; off > 0; off >>= 1) { if (t < off) red[t] = fmaxf(red[t], red[t + off]); __syncthreads(); }
  m = red[0]; __syncthreads();
  float sum = 0.0f;
  for (int i = t; i < NDIM; i += 256) { const float e = __expf(row[i] - m); row[i] = e; sum += e; }
  red[t] = sum; __syncthreads();
  for (int off = 128; off > 0; off >>= 1) { if (t < off) red[t] += red[t + off]; __syncthreads(); }
  const float inv = 1.0f / red[0];
  for (int i = t; i < NDIM; i += 256) row[i] *= inv;
}

// ---------------------------------------------------------------------------
// Softmax over the 32 channels of each head, per (b,h,n), in place
// ---------------------------------------------------------------------------
__global__ __launch_bounds__(256) void softmax_chan_kernel(float* __restrict__ q)
{
  const int g = blockIdx.x * 256 + threadIdx.x;    // B*HEADS*N threads
  const int n = g & (NDIM - 1);
  const int h = (g >> 12) & 7;
  const int b = g >> 15;
  float* base = q + ((size_t)b * CDIM + h * 32) * NDIM + n;
  float vals[32];
  float m = -1e30f;
  #pragma unroll
  for (int j = 0; j < 32; ++j) { vals[j] = base[(size_t)j * NDIM]; m = fmaxf(m, vals[j]); }
  float s = 0.0f;
  #pragma unroll
  for (int j = 0; j < 32; ++j) { vals[j] = __expf(vals[j] - m); s += vals[j]; }
  const float inv = 1.0f / s;
  #pragma unroll
  for (int j = 0; j < 32; ++j) base[(size_t)j * NDIM] = vals[j] * inv;
}

// ---------------------------------------------------------------------------
// context^T[bh][vc][kc] = sum_n k[bh,kc,n] * v[bh,vc,n]  (32x32 per head)
// WMMA over K=n; 4 waves per block, one 16x16 tile each.
// ---------------------------------------------------------------------------
__global__ __launch_bounds__(128) void context_kernel(
    const float* __restrict__ ksm, const float* __restrict__ v,
    float* __restrict__ ctxT)
{
  const int bh = blockIdx.x;                       // b*8+h
  const int b = bh >> 3, h = bh & 7;
  const float* kbase = ksm + ((size_t)b * CDIM + h * 32) * NDIM;
  const float* vbase = v   + ((size_t)b * CDIM + h * 32) * NDIM;
  const int tid = threadIdx.x, wid = tid >> 5, lane = tid & 31;
  const int half = lane >> 4, lr = lane & 15;
  const int kc0 = (wid & 1) * 16, vc0 = (wid >> 1) * 16;
  v8f acc = {};
  for (int n = 0; n < NDIM; n += 32) {
    const float* arow = kbase + (size_t)(kc0 + lr) * NDIM + n + half * 8;
    const float4 a0 = *(const float4*)(arow);
    const float4 a1 = *(const float4*)(arow + 4);
    const float4 a2 = *(const float4*)(arow + 16);
    const float4 a3 = *(const float4*)(arow + 20);
    const float* brow = vbase + (size_t)(vc0 + lr) * NDIM + n + half * 8;
    const float4 b0 = *(const float4*)(brow);
    const float4 b1 = *(const float4*)(brow + 4);
    const float4 b2 = *(const float4*)(brow + 16);
    const float4 b3 = *(const float4*)(brow + 20);
    V16 a, bf_;
    a.v[0]=(bf16)a0.x; a.v[1]=(bf16)a0.y; a.v[2]=(bf16)a0.z; a.v[3]=(bf16)a0.w;
    a.v[4]=(bf16)a1.x; a.v[5]=(bf16)a1.y; a.v[6]=(bf16)a1.z; a.v[7]=(bf16)a1.w;
    a.v[8]=(bf16)a2.x; a.v[9]=(bf16)a2.y; a.v[10]=(bf16)a2.z; a.v[11]=(bf16)a2.w;
    a.v[12]=(bf16)a3.x; a.v[13]=(bf16)a3.y; a.v[14]=(bf16)a3.z; a.v[15]=(bf16)a3.w;
    bf_.v[0]=(bf16)b0.x; bf_.v[1]=(bf16)b0.y; bf_.v[2]=(bf16)b0.z; bf_.v[3]=(bf16)b0.w;
    bf_.v[4]=(bf16)b1.x; bf_.v[5]=(bf16)b1.y; bf_.v[6]=(bf16)b1.z; bf_.v[7]=(bf16)b1.w;
    bf_.v[8]=(bf16)b2.x; bf_.v[9]=(bf16)b2.y; bf_.v[10]=(bf16)b2.z; bf_.v[11]=(bf16)b2.w;
    bf_.v[12]=(bf16)b3.x; bf_.v[13]=(bf16)b3.y; bf_.v[14]=(bf16)b3.z; bf_.v[15]=(bf16)b3.w;
    acc = __builtin_amdgcn_wmma_f32_16x16x32_bf16(
        false, a.v, false, bf_.v, (short)0, acc, false, false);
  }
  float* out = ctxT + (size_t)bh * 32 * 32;
  const int vc = vc0 + lr;
  #pragma unroll
  for (int r = 0; r < 8; ++r) {
    const int kc = kc0 + half * 8 + r;
    out[vc * 32 + kc] = acc[r];                    // store transposed
  }
}

// ---------------------------------------------------------------------------
// mod[bh,vc,n] = gate[b,h*32+vc] * sum_kc ctxT[vc,kc] * q[bh,kc,n]
// Single-K-step WMMA (K=32). Block = 8 waves: 2 vc-halves x 4 n-tiles.
// ---------------------------------------------------------------------------
__global__ __launch_bounds__(256) void attended_kernel(
    const float* __restrict__ ctxT, const float* __restrict__ qsm,
    const float* __restrict__ gate, float* __restrict__ mod)
{
  const int bh = blockIdx.y;
  const int b = bh >> 3, h = bh & 7;
  const int nblk = blockIdx.x * 64;
  __shared__ bf16 cs[32][40];                      // ctxT[vc][kc] in bf16
  const int tid = threadIdx.x;
  const float* cp = ctxT + (size_t)bh * 1024;
  for (int i = tid; i < 1024; i += 256) cs[i >> 5][i & 31] = (bf16)cp[i];
  __syncthreads();
  const int wid = tid >> 5, lane = tid & 31, half = lane >> 4, lr = lane & 15;
  const int mh = wid & 1;                          // which 16-row vc half
  const int n0 = nblk + (wid >> 1) * 16;
  V16 a;
  const bf16* ap = &cs[mh * 16 + lr][half * 8];
  a.h[0] = *(const v8bf*)ap;
  a.h[1] = *(const v8bf*)(ap + 16);
  V16 bf_;                                         // B: q column gather (coalesced per j)
  const float* qb = qsm + ((size_t)b * CDIM + h * 32 + half * 8) * NDIM + n0 + lr;
  #pragma unroll
  for (int j = 0; j < 16; ++j) bf_.v[j] = (bf16)qb[(size_t)(j + (j >> 3) * 8) * NDIM];
  v8f acc = {};
  acc = __builtin_amdgcn_wmma_f32_16x16x32_bf16(
      false, a.v, false, bf_.v, (short)0, acc, false, false);
  #pragma unroll
  for (int r = 0; r < 8; ++r) {
    const int vc = mh * 16 + half * 8 + r;
    const float gv = gate[b * CDIM + h * 32 + vc];
    mod[((size_t)b * CDIM + h * 32 + vc) * NDIM + n0 + lr] = acc[r] * gv;
  }
}

// ---------------------------------------------------------------------------
extern "C" void kernel_launch(void* const* d_in, const int* in_sizes, int n_in,
                              void* d_out, int out_size, void* d_ws, size_t ws_size,
                              hipStream_t stream) {
  const float* x      = (const float*)d_in[0];
  const float* proto  = (const float*)d_in[1];
  const float* qsw    = (const float*)d_in[2];
  const float* qsb    = (const float*)d_in[3];
  const float* qlw    = (const float*)d_in[4];
  const float* qlb    = (const float*)d_in[5];
  const float* ksw    = (const float*)d_in[6];
  const float* ksb    = (const float*)d_in[7];
  const float* klw    = (const float*)d_in[8];
  const float* klb    = (const float*)d_in[9];
  const float* wq     = (const float*)d_in[10];
  const float* bq     = (const float*)d_in[11];
  const float* wk     = (const float*)d_in[12];
  const float* bk     = (const float*)d_in[13];
  const float* wv     = (const float*)d_in[14];
  const float* bv     = (const float*)d_in[15];
  const float* mod_w  = (const float*)d_in[16];
  const float* mod_b  = (const float*)d_in[17];
  const float* ln_g   = (const float*)d_in[18];
  const float* ln_b   = (const float*)d_in[19];
  const float* wr     = (const float*)d_in[20];
  const float* br     = (const float*)d_in[21];

  const size_t BIG = (size_t)BSZ * CDIM * NDIM;    // 8388608 floats
  float* gate = (float*)d_ws;                      // [8][256]
  float* ctx  = gate + 2048;                       // [64][32][32]
  float* bufA = ctx + 65536;
  float* bufB = bufA + BIG;
  float* bufC = bufB + BIG;
  float* bufD = bufC + BIG;

  // 1) support gate
  gate_kernel<<<BSZ, 256, 0, stream>>>(proto, mod_w, mod_b, ln_g, ln_b, gate);
  // 2) fused depthwise chains: qp -> bufA, kp -> bufB
  dw_fused_kernel<<<BSZ * CDIM, 256, 0, stream>>>(x, qsw, qsb, qlw, qlb, bufA);
  dw_fused_kernel<<<BSZ * CDIM, 256, 0, stream>>>(x, ksw, ksb, klw, klb, bufB);
  // 3) projection GEMMs (WMMA): q->bufC, k->bufD, v->bufA (A free after q GEMM)
  dim3 gg(NDIM / 128, CDIM / 128, BSZ);
  gemm_wmma_kernel<<<gg, 256, 0, stream>>>(wq, bq, bufA, bufC);
  gemm_wmma_kernel<<<gg, 256, 0, stream>>>(wk, bk, bufB, bufD);
  gemm_wmma_kernel<<<gg, 256, 0, stream>>>(wv, bv, x, bufA);
  // 4) softmaxes in place
  softmax_chan_kernel<<<(BSZ * HEADS * NDIM) / 256, 256, 0, stream>>>(bufC);
  softmax_rows_kernel<<<BSZ * CDIM, 256, 0, stream>>>(bufD);
  // 5) per-head context (k x v^T), stored transposed
  context_kernel<<<BSZ * HEADS, 128, 0, stream>>>(bufD, bufA, ctx);
  // 6) attended = ctx^T x q, scaled by gate -> bufB
  attended_kernel<<<dim3(NDIM / 64, BSZ * HEADS), 256, 0, stream>>>(ctx, bufC, gate, bufB);
  // 7) output projection (WMMA) -> d_out
  gemm_wmma_kernel<<<gg, 256, 0, stream>>>(wr, br, bufB, (float*)d_out);
}
